// StructureMapping_44796508897482
// MI455X (gfx1250) — matrix operand order repaired and checked by reading, not compiled, not run on hardware
//
#include <hip/hip_runtime.h>
#include <hip/hip_bf16.h>

typedef _Float16 v16h  __attribute__((ext_vector_type(16)));
typedef _Float16 f16x8 __attribute__((ext_vector_type(8)));
typedef float    v8f   __attribute__((ext_vector_type(8)));

#define EDIM 256
#define HDIM 256
#define NB   2
#define NN   20
#define RR   400      // NN*NN
#define MROWS 800     // NB*NN*NN

__device__ inline float wave_sum_f(float v) {
#pragma unroll
  for (int off = 16; off > 0; off >>= 1) v += __shfl_xor(v, off, 32);
  return v;
}
__device__ inline float wave_max_f(float v) {
#pragma unroll
  for (int off = 16; off > 0; off >>= 1) v = fmaxf(v, __shfl_xor(v, off, 32));
  return v;
}

// ---------------------------------------------------------------------------
// f32 -> f16 transposed weight copies (so WMMA B-fragments are contiguous).
// Wt[n*K + k] = W[k*N + n]
// ---------------------------------------------------------------------------
__global__ void prep_weights_kernel(const float* __restrict__ W2,
                                    const float* __restrict__ Wm1,
                                    _Float16* __restrict__ Wt2,
                                    _Float16* __restrict__ Wm1aT,
                                    _Float16* __restrict__ Wm1bT) {
  int n = blockIdx.x;   // output row = original column
  int k = threadIdx.x;  // original row
  const float* src;
  _Float16* dst;
  if (blockIdx.y == 0)      { src = W2;               dst = Wt2;   }
  else if (blockIdx.y == 1) { src = Wm1;              dst = Wm1aT; }
  else                      { src = Wm1 + HDIM*HDIM;  dst = Wm1bT; }
  dst[n * HDIM + k] = (_Float16)src[k * HDIM + n];
}

// ---------------------------------------------------------------------------
// Six 40x256x256 matvec-GEMMs: hi/hj (W1 halves) for src/tgt, ca/cb (Wc1).
// ---------------------------------------------------------------------------
__global__ void row_gemm_kernel(const float* __restrict__ srcEnt,
                                const float* __restrict__ tgtEnt,
                                const float* __restrict__ W1,
                                const float* __restrict__ Wc1,
                                float* hi_s, float* hj_s,
                                float* hi_t, float* hj_t,
                                float* ca,   float* cb) {
  int row = blockIdx.x;   // 0..39  (b*NN + i)
  int sel = blockIdx.y;   // 0..5
  int t   = threadIdx.x;  // output column
  const float* ent; const float* W; float* out;
  switch (sel) {
    case 0:  ent = srcEnt; W = W1;               out = hi_s; break;
    case 1:  ent = srcEnt; W = W1  + EDIM*HDIM;  out = hj_s; break;
    case 2:  ent = tgtEnt; W = W1;               out = hi_t; break;
    case 3:  ent = tgtEnt; W = W1  + EDIM*HDIM;  out = hj_t; break;
    case 4:  ent = srcEnt; W = Wc1;              out = ca;   break;
    default: ent = tgtEnt; W = Wc1 + EDIM*HDIM;  out = cb;   break;
  }
  const float* er = ent + row * EDIM;
  float acc = 0.f;
#pragma unroll 4
  for (int k = 0; k < EDIM; ++k) acc = fmaf(er[k], W[k * HDIM + t], acc);
  out[row * HDIM + t] = acc;
}

// ---------------------------------------------------------------------------
// X[r,:] = f16(relu(LN(hi[b,i]+hj[b,j]+b1)*g1 + be1)),  r = b*400 + i*20 + j
// ---------------------------------------------------------------------------
__global__ void build_x_kernel(const float* __restrict__ hi,
                               const float* __restrict__ hj,
                               const float* __restrict__ b1,
                               const float* __restrict__ g1,
                               const float* __restrict__ be1,
                               _Float16* __restrict__ X) {
  __shared__ float red[256];
  __shared__ float red2[256];
  int r   = blockIdx.x;
  int b   = r / RR;
  int rem = r - b * RR;
  int i   = rem / NN;
  int j   = rem - i * NN;
  int t   = threadIdx.x;
  float pre = hi[(b*NN + i) * HDIM + t] + hj[(b*NN + j) * HDIM + t] + b1[t];
  red[t] = pre; red2[t] = pre * pre;
  __syncthreads();
  for (int s = 128; s > 0; s >>= 1) {
    if (t < s) { red[t] += red[t+s]; red2[t] += red2[t+s]; }
    __syncthreads();
  }
  float mean = red[0]  * (1.f / HDIM);
  float var  = red2[0] * (1.f / HDIM) - mean * mean;
  float inv  = rsqrtf(var + 1e-5f);
  float x    = (pre - mean) * inv * g1[t] + be1[t];
  X[r * HDIM + t] = (_Float16)fmaxf(x, 0.f);
}

// ---------------------------------------------------------------------------
// 800x256x256 GEMM with v_wmma_f32_16x16x32_f16.
// Block = 8 waves; tile 16(M) x 128(N); grid (50, 2).
// mode 0: out = f16( (acc + bias[col]) * (rel[row]>0) )
// mode 1: out = f32( acc + (bias? bias[col] : 0) )
// ---------------------------------------------------------------------------
__global__ void wmma_gemm_kernel(const _Float16* __restrict__ A,   // [800,256] row-major
                                 const _Float16* __restrict__ Bt,  // [256(n),256(k)]
                                 const float* __restrict__ bias,   // [256] or null
                                 const int* __restrict__ rel,      // [800] or null
                                 float* __restrict__ outF,
                                 _Float16* __restrict__ outH,
                                 int mode) {
  const int K = 256, NC = 256;
  int lane = threadIdx.x & 31;
  int wave = threadIdx.x >> 5;
  int row0 = blockIdx.x * 16;
  int n0   = blockIdx.y * 128 + wave * 16;

  // A fragment: lane<16 -> row M=lane, K = {0..7,16..23}; lane>=16 -> K+8.
  int m  = lane & 15;
  int kb = (lane >> 4) * 8;
  // B fragment: lane -> column n0+(lane&15), K = {0..7,16..23}; lane>=16 -> K+8.
  int bks = (lane >> 4) * 8;

  const _Float16* arow = A  + (size_t)(row0 + m) * K + kb;
  const _Float16* brow = Bt + (size_t)(n0 + (lane & 15)) * K + bks;

  v8f acc = {};
#pragma unroll
  for (int k0 = 0; k0 < K; k0 += 32) {
    union { v16h v; f16x8 h[2]; } ua, ub;
    ua.h[0] = *(const f16x8*)(arow + k0);
    ua.h[1] = *(const f16x8*)(arow + k0 + 16);
    ub.h[0] = *(const f16x8*)(brow + k0);
    ub.h[1] = *(const f16x8*)(brow + k0 + 16);
    acc = __builtin_amdgcn_wmma_f32_16x16x32_f16(false, ua.v, false, ub.v,
                                                 (short)0, acc, false, false);
  }

  // D layout: VGPR r -> M = r (lanes 0-15) / r+8 (lanes 16-31), col = lane&15.
  int col   = n0 + (lane & 15);
  int rbase = row0 + ((lane >> 4) << 3);
  float bv  = bias ? bias[col] : 0.f;
  if (mode == 0) {
#pragma unroll
    for (int r = 0; r < 8; ++r) {
      int row = rbase + r;
      float mk = (rel[row] > 0) ? 1.f : 0.f;
      outH[(size_t)row * NC + col] = (_Float16)((acc[r] + bv) * mk);
    }
  } else {
#pragma unroll
    for (int r = 0; r < 8; ++r) {
      int row = rbase + r;
      outF[(size_t)row * NC + col] = acc[r] + bv;
    }
  }
}

// ---------------------------------------------------------------------------
// One block per (b,s): 8 waves x 50 t-iterations; per-pair LN+ReLU+dot(Wm2);
// fused softmax over the 400 t's, writes final probabilities.
// sa already contains +bm1.
// ---------------------------------------------------------------------------
__global__ void mapping_rows_kernel(const float* __restrict__ sa,  // [B*R, H]
                                    const float* __restrict__ tb,  // [B*R, H]
                                    const float* __restrict__ gm,
                                    const float* __restrict__ bem,
                                    const float* __restrict__ wm2,
                                    const float* __restrict__ bm2,
                                    float* __restrict__ out) {
  __shared__ float sc[RR];
  __shared__ float red[256];
  int bs   = blockIdx.x;       // 0..799 = b*400 + s
  int b    = bs / RR;
  int lane = threadIdx.x & 31;
  int wave = threadIdx.x >> 5;
  int h0   = lane * 8;

  const float* sar = sa + (size_t)bs * HDIM + h0;
  float sa8[8], gm8[8], be8[8], w28[8];
#pragma unroll
  for (int q = 0; q < 8; ++q) {
    sa8[q] = sar[q];
    gm8[q] = gm[h0 + q];
    be8[q] = bem[h0 + q];
    w28[q] = wm2[h0 + q];
  }
  float bm2v = bm2[0];

  for (int it = 0; it < RR / 8; ++it) {
    int t = it * 8 + wave;
    const float* tbr = tb + ((size_t)b * RR + t) * HDIM + h0;
    float pre[8];
    float sm = 0.f, sq = 0.f;
#pragma unroll
    for (int q = 0; q < 8; ++q) {
      float p = sa8[q] + tbr[q];
      pre[q] = p; sm += p; sq = fmaf(p, p, sq);
    }
    sm = wave_sum_f(sm);
    sq = wave_sum_f(sq);
    float mean = sm * (1.f / HDIM);
    float var  = sq * (1.f / HDIM) - mean * mean;
    float inv  = rsqrtf(var + 1e-5f);
    float a = 0.f;
#pragma unroll
    for (int q = 0; q < 8; ++q) {
      float x = (pre[q] - mean) * inv * gm8[q] + be8[q];
      a = fmaf(fmaxf(x, 0.f), w28[q], a);
    }
    a = wave_sum_f(a);
    if (lane == 0) sc[t] = a + bm2v;
  }
  __syncthreads();

  // softmax over 400 entries with 256 threads (each owns tid and tid+256)
  int tid = threadIdx.x;
  float lm = sc[tid];
  if (tid + 256 < RR) lm = fmaxf(lm, sc[tid + 256]);
  red[tid] = lm;
  __syncthreads();
  for (int s = 128; s > 0; s >>= 1) {
    if (tid < s) red[tid] = fmaxf(red[tid], red[tid + s]);
    __syncthreads();
  }
  float mx = red[0];
  __syncthreads();
  float e0 = __expf(sc[tid] - mx);
  float e1 = (tid + 256 < RR) ? __expf(sc[tid + 256] - mx) : 0.f;
  red[tid] = e0 + e1;
  __syncthreads();
  for (int s = 128; s > 0; s >>= 1) {
    if (tid < s) red[tid] += red[tid + s];
    __syncthreads();
  }
  float invs = 1.f / red[0];
  size_t base = (size_t)bs * RR;
  out[base + tid] = e0 * invs;
  if (tid + 256 < RR) out[base + tid + 256] = e1 * invs;
}

// ---------------------------------------------------------------------------
// One wave per (b,i): relu(ca+cb+bc1) . Wc2, fused softmax over j (20).
// ---------------------------------------------------------------------------
__global__ void corr_rows_kernel(const float* __restrict__ ca,
                                 const float* __restrict__ cb,
                                 const float* __restrict__ bc1,
                                 const float* __restrict__ wc2,
                                 const float* __restrict__ bc2,
                                 float* __restrict__ out) {
  __shared__ float sc[NN];
  int blk  = blockIdx.x;        // 0..39 = b*NN + i
  int b    = blk / NN;
  int lane = threadIdx.x;       // 32 threads
  int h0   = lane * 8;

  const float* car = ca + (size_t)blk * HDIM + h0;
  float cav[8], w8[8];
#pragma unroll
  for (int q = 0; q < 8; ++q) { cav[q] = car[q] + bc1[h0 + q]; w8[q] = wc2[h0 + q]; }
  float bc2v = bc2[0];

  for (int j = 0; j < NN; ++j) {
    const float* cbr = cb + (size_t)(b * NN + j) * HDIM + h0;
    float a = 0.f;
#pragma unroll
    for (int q = 0; q < 8; ++q)
      a = fmaf(fmaxf(cav[q] + cbr[q], 0.f), w8[q], a);
    a = wave_sum_f(a);
    if (lane == 0) sc[j] = a + bc2v;
  }
  __syncthreads();
  float v  = (lane < NN) ? sc[lane] : -3.4e38f;
  float mx = wave_max_f(v);
  float e  = (lane < NN) ? __expf(v - mx) : 0.f;
  float sm = wave_sum_f(e);
  if (lane < NN) out[(size_t)blk * NN + lane] = e / sm;
}

// ---------------------------------------------------------------------------
extern "C" void kernel_launch(void* const* d_in, const int* in_sizes, int n_in,
                              void* d_out, int out_size, void* d_ws, size_t ws_size,
                              hipStream_t stream) {
  (void)in_sizes; (void)n_in; (void)out_size; (void)ws_size;
  const float* srcEnt = (const float*)d_in[0];
  const float* tgtEnt = (const float*)d_in[1];
  const int*   srcRel = (const int*)d_in[2];
  const int*   tgtRel = (const int*)d_in[3];
  const float* W1  = (const float*)d_in[4];
  const float* b1  = (const float*)d_in[5];
  const float* g1  = (const float*)d_in[6];
  const float* be1 = (const float*)d_in[7];
  const float* W2  = (const float*)d_in[8];
  const float* b2  = (const float*)d_in[9];
  const float* Wm1 = (const float*)d_in[10];
  const float* bm1 = (const float*)d_in[11];
  const float* gm  = (const float*)d_in[12];
  const float* bem = (const float*)d_in[13];
  const float* Wm2 = (const float*)d_in[14];
  const float* bm2 = (const float*)d_in[15];
  const float* Wc1 = (const float*)d_in[16];
  const float* bc1 = (const float*)d_in[17];
  const float* Wc2 = (const float*)d_in[18];
  const float* bc2 = (const float*)d_in[19];

  char* ws = (char*)d_ws;
  size_t off = 0;
  auto alloc = [&](size_t bytes) -> void* {
    void* p = ws + off;
    off += (bytes + 255) & ~(size_t)255;
    return p;
  };

  float* hi_s = (float*)alloc(40 * 256 * 4);
  float* hj_s = (float*)alloc(40 * 256 * 4);
  float* hi_t = (float*)alloc(40 * 256 * 4);
  float* hj_t = (float*)alloc(40 * 256 * 4);
  float* caB  = (float*)alloc(40 * 256 * 4);
  float* cbB  = (float*)alloc(40 * 256 * 4);
  float* sa   = (float*)alloc((size_t)MROWS * 256 * 4);
  float* tbB  = (float*)alloc((size_t)MROWS * 256 * 4);
  _Float16* Xs    = (_Float16*)alloc((size_t)MROWS * 256 * 2);
  _Float16* Xt    = (_Float16*)alloc((size_t)MROWS * 256 * 2);
  _Float16* Rs    = (_Float16*)alloc((size_t)MROWS * 256 * 2);
  _Float16* Rt    = (_Float16*)alloc((size_t)MROWS * 256 * 2);
  _Float16* Wt2   = (_Float16*)alloc(256 * 256 * 2);
  _Float16* Wm1aT = (_Float16*)alloc(256 * 256 * 2);
  _Float16* Wm1bT = (_Float16*)alloc(256 * 256 * 2);

  // 1) f16 transposed weights for WMMA B operands
  prep_weights_kernel<<<dim3(256, 3), 256, 0, stream>>>(W2, Wm1, Wt2, Wm1aT, Wm1bT);
  // 2) hi/hj/ca/cb matvec-GEMMs (40 rows each)
  row_gemm_kernel<<<dim3(40, 6), 256, 0, stream>>>(srcEnt, tgtEnt, W1, Wc1,
                                                   hi_s, hj_s, hi_t, hj_t, caB, cbB);
  // 3) LN+ReLU of broadcast sum -> f16 A-matrices
  build_x_kernel<<<800, 256, 0, stream>>>(hi_s, hj_s, b1, g1, be1, Xs);
  build_x_kernel<<<800, 256, 0, stream>>>(hi_t, hj_t, b1, g1, be1, Xt);
  // 4) X @ W2 + b2, masked -> src_rel/tgt_rel (f16)   [WMMA]
  wmma_gemm_kernel<<<dim3(50, 2), 256, 0, stream>>>(Xs, Wt2, b2, srcRel, nullptr, Rs, 0);
  wmma_gemm_kernel<<<dim3(50, 2), 256, 0, stream>>>(Xt, Wt2, b2, tgtRel, nullptr, Rt, 0);
  // 5) sa = src_rel @ Wm1[:H] + bm1 ; tb = tgt_rel @ Wm1[H:]   [WMMA]
  wmma_gemm_kernel<<<dim3(50, 2), 256, 0, stream>>>(Rs, Wm1aT, bm1, nullptr, sa, nullptr, 1);
  wmma_gemm_kernel<<<dim3(50, 2), 256, 0, stream>>>(Rt, Wm1bT, nullptr, nullptr, tbB, nullptr, 1);
  // 6) pairwise LN+ReLU+dot(Wm2) + fused softmax over t -> d_out[0:320000]
  mapping_rows_kernel<<<800, 256, 0, stream>>>(sa, tbB, gm, bem, Wm2, bm2, (float*)d_out);
  // 7) correspondence scores + fused softmax over j -> d_out[320000:320800]
  corr_rows_kernel<<<40, 32, 0, stream>>>(caB, cbB, bc1, Wc2, bc2, (float*)d_out + 320000);
}
